// GAT_79173427134539
// MI455X (gfx1250) — matrix operand order repaired and checked by reading, not compiled, not run on hardware
//
#include <hip/hip_runtime.h>
#include <hip/hip_bf16.h>
#include <math.h>

#define H_IMG 512
#define W_IMG 512
#define N_PTS 262144
#define N_EDGES 2097152
#define N_NODES (N_PTS + H_IMG * W_IMG)   // 524288
#define IMG_ELEMS (16 * H_IMG * W_IMG)    // final layer hc=16 -> 4194304
#define NEG_SLOPE 0.2f

typedef __attribute__((ext_vector_type(16))) __bf16 v16bf;
typedef __attribute__((ext_vector_type(8)))  float  v8f;

// ---- monotonic float <-> uint mapping for atomicMax-based segment_max ----
__device__ __forceinline__ unsigned fenc(float f) {
  unsigned u = __float_as_uint(f);
  return (u >> 31) ? ~u : (u | 0x80000000u);
}
__device__ __forceinline__ float fdec(unsigned u) {
  unsigned v = (u >> 31) ? (u & 0x7fffffffu) : ~u;
  return __uint_as_float(v);
}
__device__ __forceinline__ float leaky(float z) { return z > 0.f ? z : NEG_SLOPE * z; }

// ---- layer-0 features, point half: contiguous copy+cast ----
__global__ void prep_ptx_kernel(const float* __restrict__ ptx, __bf16* __restrict__ A) {
  int i = blockIdx.x * blockDim.x + threadIdx.x;          // over N_PTS*64
  if (i >= N_PTS * 64) return;
  A[i] = (__bf16)ptx[i];
}

// ---- layer-0 features, image half: CHW -> (pixel, channel) via LDS transpose ----
__global__ void prep_img_kernel(const float* __restrict__ x, __bf16* __restrict__ A) {
  __shared__ float t[64][65];
  int p0 = blockIdx.x * 64;                                // 4096 blocks
  for (int idx = threadIdx.x; idx < 64 * 64; idx += 256) {
    int c = idx >> 6, pp = idx & 63;                       // coalesced over pixels
    t[c][pp] = x[(size_t)c * (H_IMG * W_IMG) + p0 + pp];
  }
  __syncthreads();
  for (int idx = threadIdx.x; idx < 64 * 64; idx += 256) {
    int pp = idx >> 6, c = idx & 63;                       // coalesced over channels
    A[(size_t)(N_PTS + p0 + pp) * 64 + c] = (__bf16)t[c][pp];
  }
}

// ---- h = A(bf16) @ W, v_wmma_f32_16x16x32_bf16, B pre-shuffled in LDS ----
template <int CIN, int HC>
__global__ void gemm_kernel(const __bf16* __restrict__ A, const float* __restrict__ Wf,
                            float* __restrict__ Hf) {
  constexpr int KT = CIN / 32;                 // k tiles
  constexpr int NT = HC / 16;                  // n tiles
  __shared__ alignas(16) __bf16 Wl[KT * NT * 512];

  int tid = threadIdx.x;
  // Stage W directly in per-lane fragment order: tile-major, lane's 16 values contiguous.
  for (int idx = tid; idx < CIN * HC; idx += 256) {
    int tile = idx >> 9, r = idx & 511;
    int lane = r >> 4, i = r & 15;
    int kt = tile / NT, nt = tile - kt * NT;
    int K = kt * 32 + (lane >> 4) * 8 + (i & 7) + (i >> 3) * 16;    // B: lanes 0-15 K=0..7,16..23; 16-31 K=8..15,24..31
    int N = nt * 16 + (lane & 15);
    Wl[idx] = (__bf16)Wf[K * HC + N];
  }
  __syncthreads();

  int wave = tid >> 5, lane = tid & 31;
  int mIdx = lane & 15, hi = lane >> 4;
  int rowBase = blockIdx.x * 128 + wave * 16;
  const __bf16* ap = A + (size_t)(rowBase + mIdx) * CIN;

  union BC { v16bf v; uint4 u[2]; };
  BC afr[KT];
  #pragma unroll
  for (int kt = 0; kt < KT; kt++) {            // A: two contiguous 16B runs per lane
    int kb = kt * 32 + hi * 8;
    afr[kt].u[0] = *(const uint4*)(ap + kb);
    afr[kt].u[1] = *(const uint4*)(ap + kb + 16);
  }
  #pragma unroll
  for (int nt = 0; nt < NT; nt++) {
    v8f acc = {};
    #pragma unroll
    for (int kt = 0; kt < KT; kt++) {
      BC bfr;
      const uint4* bp = (const uint4*)(Wl + ((kt * NT + nt) << 9) + lane * 16);
      bfr.u[0] = bp[0];
      bfr.u[1] = bp[1];
      acc = __builtin_amdgcn_wmma_f32_16x16x32_bf16(false, afr[kt].v, false, bfr.v,
                                                    (short)0, acc, false, false);
    }
    // D: lane holds column N=mIdx; VGPR r -> row M = r + hi*8
    float* hp = Hf + (size_t)rowBase * HC + nt * 16 + mIdx;
    #pragma unroll
    for (int r = 0; r < 8; r++) hp[(size_t)(r + hi * 8) * HC] = acc[r];
  }
}

// ---- per-node attention logit halves ----
template <int HEADS, int CH>
__global__ void alpha_kernel(const float* __restrict__ Hf, const float* __restrict__ awS,
                             const float* __restrict__ awD, float* __restrict__ aS,
                             float* __restrict__ aD) {
  int node = blockIdx.x * blockDim.x + threadIdx.x;
  if (node >= N_NODES) return;
  constexpr int HCv = HEADS * CH;
  const float4* hp = (const float4*)(Hf + (size_t)node * HCv);
  float hv[HCv];
  #pragma unroll
  for (int q = 0; q < HCv / 4; q++) {
    float4 f = hp[q];
    hv[4 * q] = f.x; hv[4 * q + 1] = f.y; hv[4 * q + 2] = f.z; hv[4 * q + 3] = f.w;
  }
  #pragma unroll
  for (int h = 0; h < HEADS; h++) {
    float ds = 0.f, dd = 0.f;
    #pragma unroll
    for (int c = 0; c < CH; c++) {
      ds += hv[h * CH + c] * awS[h * CH + c];
      dd += hv[h * CH + c] * awD[h * CH + c];
    }
    aS[node * HEADS + h] = ds;
    aD[node * HEADS + h] = dd;
  }
}

// ---- zero accumulators (capture-safe) ----
__global__ void init_kernel(float* __restrict__ Out, unsigned* __restrict__ m,
                            float* __restrict__ s, int nOut, int nMS) {
  int i = blockIdx.x * blockDim.x + threadIdx.x;
  if (i < nOut) Out[i] = 0.f;
  if (i < nMS) { m[i] = 0u; s[i] = 0.f; }     // enc-space 0 == -inf
}

// ---- segment_max over dst ----
template <int HEADS>
__global__ void edge_max_kernel(const int* __restrict__ pe, const float* __restrict__ aS,
                                const float* __restrict__ aD, unsigned* __restrict__ m) {
  int e = blockIdx.x * blockDim.x + threadIdx.x;
  if (e >= N_EDGES) return;
  int sN = pe[e], dN = pe[N_EDGES + e];
  #pragma unroll
  for (int h = 0; h < HEADS; h++) {
    float l = leaky(aS[sN * HEADS + h] + aD[dN * HEADS + h]);
    atomicMax(&m[dN * HEADS + h], fenc(l));
  }
}

// ---- segment_sum of exp(l - m[dst]) ----
template <int HEADS>
__global__ void edge_sum_kernel(const int* __restrict__ pe, const float* __restrict__ aS,
                                const float* __restrict__ aD, const unsigned* __restrict__ m,
                                float* __restrict__ s) {
  int e = blockIdx.x * blockDim.x + threadIdx.x;
  if (e >= N_EDGES) return;
  int sN = pe[e], dN = pe[N_EDGES + e];
  #pragma unroll
  for (int h = 0; h < HEADS; h++) {
    float l = leaky(aS[sN * HEADS + h] + aD[dN * HEADS + h]);
    atomicAdd(&s[dN * HEADS + h], __expf(l - fdec(m[dN * HEADS + h])));
  }
}

// ---- message aggregation: Out[dst] += alpha * h[src] ----
template <int HEADS, int CH>
__global__ void edge_agg_kernel(const int* __restrict__ pe, const float* __restrict__ aS,
                                const float* __restrict__ aD, const unsigned* __restrict__ m,
                                const float* __restrict__ s, const float* __restrict__ Hf,
                                float* __restrict__ Out) {
  int t = blockIdx.x * blockDim.x + threadIdx.x;
  if (t >= N_EDGES * HEADS) return;
  int e = t / HEADS, h = t - e * HEADS;
  int sN = pe[e], dN = pe[N_EDGES + e];
  float l = leaky(aS[sN * HEADS + h] + aD[dN * HEADS + h]);
  float alpha = __expf(l - fdec(m[dN * HEADS + h])) / (s[dN * HEADS + h] + 1e-16f);
  const float4* hp = (const float4*)(Hf + ((size_t)sN * HEADS + h) * CH);
  float* op = Out + ((size_t)dN * HEADS + h) * CH;
  #pragma unroll
  for (int q = 0; q < CH / 4; q++) {
    float4 f = hp[q];
    atomicAdd(&op[4 * q + 0], alpha * f.x);
    atomicAdd(&op[4 * q + 1], alpha * f.y);
    atomicAdd(&op[4 * q + 2], alpha * f.z);
    atomicAdd(&op[4 * q + 3], alpha * f.w);
  }
}

// ---- ELU + recast to bf16 as next layer's WMMA A operand ----
__global__ void finalize_bf_kernel(const float* __restrict__ Out, __bf16* __restrict__ A,
                                   int n) {
  int i = blockIdx.x * blockDim.x + threadIdx.x;
  if (i >= n) return;
  float v = Out[i];
  A[i] = (__bf16)(v > 0.f ? v : expm1f(v));
}

// ---- last layer: ELU + scatter into (x_img [16,H,W], ptx_out [N_PTS,16]) ----
__global__ void finalize_out_kernel(const float* __restrict__ Out, float* __restrict__ out) {
  int i = blockIdx.x * blockDim.x + threadIdx.x;     // over N_NODES*16
  if (i >= N_NODES * 16) return;
  float v = Out[i];
  v = v > 0.f ? v : expm1f(v);
  int node = i >> 4, c = i & 15;
  if (node < N_PTS) out[IMG_ELEMS + i] = v;                       // ptx_out (i == node*16+c)
  else out[(size_t)c * (H_IMG * W_IMG) + (node - N_PTS)] = v;     // x_img CHW
}

template <int CIN, int HEADS, int CH>
static void run_layer(const __bf16* A_bf, const float* Wm, const float* AwS,
                      const float* AwD, const int* pe, float* Hf, float* Out,
                      float* aS, float* aD, unsigned* mBuf, float* sBuf,
                      hipStream_t stream) {
  constexpr int HCv = HEADS * CH;
  gemm_kernel<CIN, HCv><<<N_NODES / 128, 256, 0, stream>>>(A_bf, Wm, Hf);
  alpha_kernel<HEADS, CH><<<N_NODES / 256, 256, 0, stream>>>(Hf, AwS, AwD, aS, aD);
  init_kernel<<<(N_NODES * HCv) / 256, 256, 0, stream>>>(Out, mBuf, sBuf,
                                                         N_NODES * HCv, N_NODES * HEADS);
  edge_max_kernel<HEADS><<<N_EDGES / 256, 256, 0, stream>>>(pe, aS, aD, mBuf);
  edge_sum_kernel<HEADS><<<N_EDGES / 256, 256, 0, stream>>>(pe, aS, aD, mBuf, sBuf);
  edge_agg_kernel<HEADS, CH><<<(N_EDGES * HEADS) / 256, 256, 0, stream>>>(pe, aS, aD, mBuf,
                                                                          sBuf, Hf, Out);
}

extern "C" void kernel_launch(void* const* d_in, const int* in_sizes, int n_in,
                              void* d_out, int out_size, void* d_ws, size_t ws_size,
                              hipStream_t stream) {
  const float* x   = (const float*)d_in[0];
  const float* ptx = (const float*)d_in[1];
  const int*   pe  = (const int*)d_in[5];
  const float* W0  = (const float*)d_in[6];
  const float* aS0 = (const float*)d_in[7];
  const float* aD0 = (const float*)d_in[8];
  const float* W1  = (const float*)d_in[9];
  const float* aS1 = (const float*)d_in[10];
  const float* aD1 = (const float*)d_in[11];
  const float* W2  = (const float*)d_in[12];
  const float* aS2 = (const float*)d_in[13];
  const float* aD2 = (const float*)d_in[14];
  float* out = (float*)d_out;

  char* ws = (char*)d_ws;
  __bf16*   A_bf = (__bf16*)(ws);                                   // 67,108,864 B
  float*    Hf   = (float*)(ws + 67108864);                         // 67,108,864 B
  float*    Out  = (float*)(ws + 134217728);                        // 67,108,864 B
  float*    aS   = (float*)(ws + 201326592);                        //  8,388,608 B
  float*    aD   = (float*)(ws + 209715200);                        //  8,388,608 B
  unsigned* mBuf = (unsigned*)(ws + 218103808);                     //  8,388,608 B
  float*    sBuf = (float*)(ws + 226492416);                        //  8,388,608 B

  prep_ptx_kernel<<<(N_PTS * 64) / 256, 256, 0, stream>>>(ptx, A_bf);
  prep_img_kernel<<<(H_IMG * W_IMG) / 64, 256, 0, stream>>>(x, A_bf);

  // Layer 0: cin=64, heads=4, ch=8
  run_layer<64, 4, 8>(A_bf, W0, aS0, aD0, pe, Hf, Out, aS, aD, mBuf, sBuf, stream);
  finalize_bf_kernel<<<(N_NODES * 32) / 256, 256, 0, stream>>>(Out, A_bf, N_NODES * 32);

  // Layer 1: cin=32, heads=4, ch=8
  run_layer<32, 4, 8>(A_bf, W1, aS1, aD1, pe, Hf, Out, aS, aD, mBuf, sBuf, stream);
  finalize_bf_kernel<<<(N_NODES * 32) / 256, 256, 0, stream>>>(Out, A_bf, N_NODES * 32);

  // Layer 2: cin=32, heads=1, ch=16
  run_layer<32, 1, 16>(A_bf, W2, aS2, aD2, pe, Hf, Out, aS, aD, mBuf, sBuf, stream);
  finalize_out_kernel<<<(N_NODES * 16) / 256, 256, 0, stream>>>(Out, out);
}
